// CulturalAttention_49546742726799
// MI455X (gfx1250) — matrix operand order repaired and checked by reading, not compiled, not run on hardware
//
#include <hip/hip_runtime.h>
#include <hip/hip_bf16.h>

// CulturalAttention for MI455X (gfx1250, wave32, WMMA + TDM double-buffered).
//
// Math note: the "cultural bias" lam * (Q @ c_proj) is added uniformly across
// the key axis for each query row; softmax over keys is invariant to per-row
// constants, so the output equals softmax(Q K^T / sqrt(dk)) @ V exactly.
// Inputs cultural_embedding/WC/bC/lam therefore do not affect the result.
//
// Pipeline:
//   1) qkv_proj: X[f32] -> Qh/Kh/Vh [f16] via v_wmma_f32_16x16x32_f16
//   2) flash_attn: 4 waves/block; K/V tiles staged in LDS by the Tensor Data
//      Mover with ping-pong buffers (TDM for tile j+1 overlaps compute of
//      tile j; per-wave in-order TENSORcnt lets us wait with count<=2).
//
// Workspace: 3 * B*S*64 f16 = 6 MB in d_ws.

typedef __attribute__((ext_vector_type(16))) _Float16     v16h;
typedef __attribute__((ext_vector_type(8)))  float        v8f;
typedef __attribute__((ext_vector_type(4)))  unsigned int v4u;
typedef __attribute__((ext_vector_type(8)))  int          v8i;
typedef __attribute__((ext_vector_type(4)))  int          v4i;

#define D_MODEL 1024
#define D_K     64
#define BATCH   4
#define SEQ     4096

// ---------------------------------------------------------------------------
// TDM 2-D tile load: global (row-major, f16) -> LDS.  Descriptor per ISA §8:
//   group0: count=1 | lds_addr | global_addr[56:0] | type=2
//   group1: data_size=2B, tensor_dim0/1, tile_dim0/1, tensor_dim0_stride
//   group2/3: zero (<=2D tensor)
// Issued once per wave (EXEC ignored by TENSOR ops); caller gates to one wave.
// ---------------------------------------------------------------------------
__device__ static inline void tdm_load_tile_2d(
    unsigned lds_off, const void* gptr,
    unsigned tensor_d0, unsigned tensor_d1, unsigned stride_d0,
    unsigned tile_d0, unsigned tile_d1)
{
    const unsigned long long ga = (unsigned long long)(uintptr_t)gptr;
    v4u g0;
    g0[0] = 1u;                                             // count=1 (valid user D#)
    g0[1] = lds_off;                                        // lds_addr (bytes)
    g0[2] = (unsigned)(ga & 0xFFFFFFFFull);                 // global_addr[31:0]
    g0[3] = (unsigned)((ga >> 32) & 0x01FFFFFFull)          // global_addr[56:32]
          | (2u << 30);                                     // type=2 ("image")
    v8i g1;
    g1[0] = (int)(1u << 16);                                // wg_mask=0, data_size=1 (2B)
    g1[1] = (int)((tensor_d0 & 0xFFFFu) << 16);             // tensor_dim0[15:0]
    g1[2] = (int)(((tensor_d0 >> 16) & 0xFFFFu)             // tensor_dim0[31:16]
          | ((tensor_d1 & 0xFFFFu) << 16));                 // tensor_dim1[15:0]
    g1[3] = (int)(((tensor_d1 >> 16) & 0xFFFFu)             // tensor_dim1[31:16]
          | ((tile_d0 & 0xFFFFu) << 16));                   // tile_dim0
    g1[4] = (int)(tile_d1 & 0xFFFFu);                       // tile_dim1 (tile_dim2=0)
    g1[5] = (int)stride_d0;                                 // tensor_dim0_stride[31:0]
    g1[6] = 0;                                              // stride hi, dim1_stride=0
    g1[7] = 0;
    v4i z = {0, 0, 0, 0};
#if defined(__clang_major__) && (__clang_major__ >= 23)
    v8i z8 = {0, 0, 0, 0, 0, 0, 0, 0};
    __builtin_amdgcn_tensor_load_to_lds(g0, g1, z, z, z8, 0);
#else
    __builtin_amdgcn_tensor_load_to_lds(g0, g1, z, z, 0);
#endif
}

// ---------------------------------------------------------------------------
// Fragment layouts used in this file (wave32, contiguous k = 16h + s):
//   A (16x32 f16): lane&15 = row M; hi half of wave selects K-halves
//   B (32x16 f16): lane&15 = col N; lanes 0-15 K=0..15, lanes 16-31 K=16..31 (this file's order, not ISA table)
//   C/D (16x16 f32): lane&15 = col N; v[r] = row r (lo lanes) / r+8 (hi lanes)
// ---------------------------------------------------------------------------

__global__ __launch_bounds__(32) void qkv_proj_kernel(
    const float* __restrict__ X,
    const float* __restrict__ WQ, const float* __restrict__ bQ,
    const float* __restrict__ WK, const float* __restrict__ bK,
    const float* __restrict__ WV, const float* __restrict__ bV,
    _Float16* __restrict__ Qh, _Float16* __restrict__ Kh, _Float16* __restrict__ Vh)
{
    const int lane  = threadIdx.x;
    const int ln15  = lane & 15;
    const int hi    = lane >> 4;
    const int mbase = blockIdx.x * 16;          // row tile in flattened [B*S]
    const int which = blockIdx.y >> 2;          // 0=Q 1=K 2=V
    const int nbase = (blockIdx.y & 3) * 16;    // 16-col tile of d_k=64

    const float* W    = (which == 0) ? WQ : (which == 1) ? WK : WV;
    const float* bias = (which == 0) ? bQ : (which == 1) ? bK : bV;
    _Float16*    Out  = (which == 0) ? Qh : (which == 1) ? Kh : Vh;

    const float* xrow = X + (size_t)(mbase + ln15) * D_MODEL;

    v8f acc = {};
    for (int kk = 0; kk < D_MODEL; kk += 32) {
        v16h a;
        const int ka = kk + hi * 16;
        #pragma unroll
        for (int i = 0; i < 8; ++i) a[i]     = (_Float16)xrow[ka + i];
        #pragma unroll
        for (int i = 0; i < 8; ++i) a[8 + i] = (_Float16)xrow[ka + 8 + i];

        v16h b;
        const int kb = kk + hi * 16;
        #pragma unroll
        for (int i = 0; i < 16; ++i)
            b[i] = (_Float16)W[(size_t)(kb + i) * D_K + nbase + ln15];

        acc = __builtin_amdgcn_wmma_f32_16x16x32_f16(
            false, a, false, b, (short)0, acc, false, false);
    }

    const float bn = bias[nbase + ln15];
    #pragma unroll
    for (int r = 0; r < 8; ++r) {
        const int m = mbase + r + hi * 8;
        Out[(size_t)m * D_K + nbase + ln15] = (_Float16)(acc[r] + bn);
    }
}

__global__ __launch_bounds__(128) void flash_attn_kernel(
    const _Float16* __restrict__ Qh, const _Float16* __restrict__ Kh,
    const _Float16* __restrict__ Vh, float* __restrict__ Out)
{
    // Ping-pong TDM destinations + transposed V + per-wave P bounce: 28 KB.
    __shared__ _Float16 K_lds[2][32][D_K];
    __shared__ _Float16 V_lds[2][32][D_K];
    __shared__ _Float16 VT_lds[2][D_K][32];
    __shared__ _Float16 pbuf[4][16][32];

    const int tid  = threadIdx.x;
    const int wv   = tid >> 5;                  // wave id: query tile within block
    const int lane = tid & 31;
    const int ln15 = lane & 15;
    const int hi   = lane >> 4;

    const int batch  = blockIdx.x >> 6;         // 64 blocks per batch
    const int qt     = blockIdx.x & 63;
    const int qbase  = batch * SEQ + qt * 64 + wv * 16;
    const int kvbase = batch * SEQ;

    // Q fragments: 16x64 = two 16x32 A-frags
    v16h qa0, qa1;
    {
        const _Float16* qrow = Qh + (size_t)(qbase + ln15) * D_K;
        const int ka = hi * 16;
        #pragma unroll
        for (int i = 0; i < 8; ++i) { qa0[i] = qrow[ka + i];      qa0[8 + i] = qrow[8 + ka + i]; }
        #pragma unroll
        for (int i = 0; i < 8; ++i) { qa1[i] = qrow[32 + ka + i]; qa1[8 + i] = qrow[40 + ka + i]; }
    }

    float m[8], l[8];
    v8f o0 = {}, o1 = {}, o2 = {}, o3 = {};
    #pragma unroll
    for (int r = 0; r < 8; ++r) { m[r] = -3.0e38f; l[r] = 0.0f; }

    // Prologue: kick off TDM for tile 0 into buffer 0.
    if (wv == 0) {
        tdm_load_tile_2d((unsigned)(uintptr_t)&K_lds[0][0][0],
                         Kh + (size_t)kvbase * D_K, D_K, SEQ, D_K, D_K, 32);
        tdm_load_tile_2d((unsigned)(uintptr_t)&V_lds[0][0][0],
                         Vh + (size_t)kvbase * D_K, D_K, SEQ, D_K, D_K, 32);
    }

    constexpr int NT = SEQ / 32;                // 128 key tiles of 32
    for (int j = 0; j < NT; ++j) {
        const int cur = j & 1;
        const int nxt = cur ^ 1;

        __syncthreads();   // buffer `nxt` free (consumed in iteration j-1)

        if (j + 1 < NT) {
            if (wv == 0) {
                // Prefetch tile j+1 while tile j is being computed.
                const int kb1 = (j + 1) * 32;
                tdm_load_tile_2d((unsigned)(uintptr_t)&K_lds[nxt][0][0],
                                 Kh + (size_t)(kvbase + kb1) * D_K, D_K, SEQ, D_K, D_K, 32);
                tdm_load_tile_2d((unsigned)(uintptr_t)&V_lds[nxt][0][0],
                                 Vh + (size_t)(kvbase + kb1) * D_K, D_K, SEQ, D_K, D_K, 32);
                // In-order per-wave TENSORcnt: <=2 outstanding means tile j's
                // two loads (issued last iteration) have completed.
                __builtin_amdgcn_s_wait_tensorcnt(2);
            }
        } else {
            if (wv == 0) __builtin_amdgcn_s_wait_tensorcnt(0);
        }
        __syncthreads();   // tile j visible to all waves

        // Cooperative V transpose: [key][d] -> [d][key] so B-frag reads are
        // contiguous 32B/lane DS loads instead of 2B scatters.
        {
            const int key = tid >> 2;
            const int seg = (tid & 3) * 16;
            #pragma unroll
            for (int i = 0; i < 16; ++i) VT_lds[cur][seg + i][key] = V_lds[cur][key][seg + i];
        }
        __syncthreads();

        // ---- S = Q K^T for 32 keys (two 16x16 logit tiles) ----
        v8f s0 = {}, s1 = {};
        #pragma unroll
        for (int t = 0; t < 2; ++t) {
            const _Float16* krow = &K_lds[cur][t * 16 + ln15][0];
            const int kq = hi * 16;
            v16h b0, b1;
            #pragma unroll
            for (int i = 0; i < 16; ++i) b0[i] = krow[kq + i];        // d = 0..31
            #pragma unroll
            for (int i = 0; i < 16; ++i) b1[i] = krow[32 + kq + i];   // d = 32..63
            v8f s = {};
            s = __builtin_amdgcn_wmma_f32_16x16x32_f16(false, qa0, false, b0, (short)0, s, false, false);
            s = __builtin_amdgcn_wmma_f32_16x16x32_f16(false, qa1, false, b1, (short)0, s, false, false);
            if (t == 0) s0 = s; else s1 = s;
        }

        // ---- online softmax update ----
        float mx[8];
        #pragma unroll
        for (int r = 0; r < 8; ++r) {
            s0[r] *= 0.125f;                 // 1/sqrt(64)
            s1[r] *= 0.125f;
            mx[r] = fmaxf(s0[r], s1[r]);
        }
        #pragma unroll
        for (int d = 1; d <= 8; d <<= 1) {   // row reduce across 16 striping lanes
            #pragma unroll
            for (int r = 0; r < 8; ++r)
                mx[r] = fmaxf(mx[r], __shfl_xor(mx[r], d, 16));
        }

        float alpha[8], psum[8];
        #pragma unroll
        for (int r = 0; r < 8; ++r) {
            const float mn = fmaxf(m[r], mx[r]);
            alpha[r] = __expf(m[r] - mn);
            m[r]     = mn;
            s0[r] = __expf(s0[r] - mn);
            s1[r] = __expf(s1[r] - mn);
            psum[r] = s0[r] + s1[r];
        }
        #pragma unroll
        for (int d = 1; d <= 8; d <<= 1) {
            #pragma unroll
            for (int r = 0; r < 8; ++r)
                psum[r] += __shfl_xor(psum[r], d, 16);
        }
        #pragma unroll
        for (int r = 0; r < 8; ++r) {
            l[r] = l[r] * alpha[r] + psum[r];
            o0[r] *= alpha[r]; o1[r] *= alpha[r];
            o2[r] *= alpha[r]; o3[r] *= alpha[r];
        }

        // ---- P: C-layout -> LDS row-major -> A-fragment (16x32 f16) ----
        #pragma unroll
        for (int r = 0; r < 8; ++r) {
            pbuf[wv][r + hi * 8][ln15]      = (_Float16)s0[r];
            pbuf[wv][r + hi * 8][16 + ln15] = (_Float16)s1[r];
        }
        asm volatile("s_wait_dscnt 0" ::: "memory");  // per-wave DS ordering
        v16h pa;
        {
            const int ka = hi * 16;
            #pragma unroll
            for (int i = 0; i < 8; ++i) { pa[i] = pbuf[wv][ln15][ka + i]; pa[8 + i] = pbuf[wv][ln15][8 + ka + i]; }
        }
        asm volatile("s_wait_dscnt 0" ::: "memory");

        // ---- O += P V : four 32x16 B-frags from transposed V (contiguous) ----
        v16h vb0, vb1, vb2, vb3;
        {
            const int kq = hi * 16;
            #pragma unroll
            for (int i = 0; i < 16; ++i) vb0[i] = VT_lds[cur][ 0 + ln15][kq + i];
            #pragma unroll
            for (int i = 0; i < 16; ++i) vb1[i] = VT_lds[cur][16 + ln15][kq + i];
            #pragma unroll
            for (int i = 0; i < 16; ++i) vb2[i] = VT_lds[cur][32 + ln15][kq + i];
            #pragma unroll
            for (int i = 0; i < 16; ++i) vb3[i] = VT_lds[cur][48 + ln15][kq + i];
        }
        o0 = __builtin_amdgcn_wmma_f32_16x16x32_f16(false, pa, false, vb0, (short)0, o0, false, false);
        o1 = __builtin_amdgcn_wmma_f32_16x16x32_f16(false, pa, false, vb1, (short)0, o1, false, false);
        o2 = __builtin_amdgcn_wmma_f32_16x16x32_f16(false, pa, false, vb2, (short)0, o2, false, false);
        o3 = __builtin_amdgcn_wmma_f32_16x16x32_f16(false, pa, false, vb3, (short)0, o3, false, false);
    }

    // ---- epilogue: O / l ----
    #pragma unroll
    for (int r = 0; r < 8; ++r) {
        const float invl = 1.0f / l[r];
        float* op = Out + (size_t)(qbase + r + hi * 8) * D_K + ln15;
        op[0]  = o0[r] * invl;
        op[16] = o1[r] * invl;
        op[32] = o2[r] * invl;
        op[48] = o3[r] * invl;
    }
}

extern "C" void kernel_launch(void* const* d_in, const int* in_sizes, int n_in,
                              void* d_out, int out_size, void* d_ws, size_t ws_size,
                              hipStream_t stream)
{
    const float* X  = (const float*)d_in[0];
    // d_in[1] cultural_embedding, d_in[8] WC, d_in[9] bC, d_in[10] lam:
    //   unused — per-query-row constant bias is a softmax no-op (see header).
    const float* WQ = (const float*)d_in[2];
    const float* bQ = (const float*)d_in[3];
    const float* WK = (const float*)d_in[4];
    const float* bK = (const float*)d_in[5];
    const float* WV = (const float*)d_in[6];
    const float* bV = (const float*)d_in[7];

    _Float16* Qh = (_Float16*)d_ws;
    _Float16* Kh = Qh + (size_t)BATCH * SEQ * D_K;
    _Float16* Vh = Kh + (size_t)BATCH * SEQ * D_K;

    // 1) QKV projection: 1024 M-tiles x (3 projections * 4 N-tiles)
    dim3 gA(BATCH * SEQ / 16, 12);
    qkv_proj_kernel<<<gA, 32, 0, stream>>>(X, WQ, bQ, WK, bK, WV, bV, Qh, Kh, Vh);

    // 2) Flash attention: 4 waves/block, 64 queries/block, TDM double-buffered
    flash_attn_kernel<<<BATCH * SEQ / 64, 128, 0, stream>>>(Qh, Kh, Vh, (float*)d_out);
}
